// GCM_22179211116811
// MI455X (gfx1250) — matrix operand order repaired and checked by reading, not compiled, not run on hardware
//
#include <hip/hip_runtime.h>

typedef _Float16 f16;
typedef _Float16 v16h __attribute__((ext_vector_type(16)));
typedef float    v8f  __attribute__((ext_vector_type(8)));
typedef unsigned v8u  __attribute__((ext_vector_type(8)));

// types for the async-to-LDS builtin (probe-derived: param is v4i in AS1/AS3)
typedef int v4i __attribute__((vector_size(16)));
typedef __attribute__((address_space(1))) v4i as1_v4i;
typedef __attribute__((address_space(3))) v4i as3_v4i;

#define WMMA_F16(a,b,c) __builtin_amdgcn_wmma_f32_16x16x32_f16(false,(a),false,(b),(short)0,(c),false,false)

#define ASYNC_OK __has_builtin(__builtin_amdgcn_global_load_async_to_lds_b128)

namespace {
constexpr int CB = 16, CT = 96, CN = 5000, CD = 2;
constexpr int CH = 256, DM = 192, DM2 = 384;
constexpr int NP = 5008;           // per-batch padded row count (multiple of 16)
constexpr int MT_B = NP / 16;      // 313 row-tiles per batch
constexpr int RP = CB * NP;        // 80128 padded rows total
constexpr int R1 = CB * CD * CN;   // 160000 gcn rows
constexpr float FEPS = 1e-6f;
constexpr float PIH = 1.57079632679f;
}

__device__ __forceinline__ v8f vzero() {
  v8f z;
#pragma unroll
  for (int i = 0; i < 8; ++i) z[i] = 0.f;
  return z;
}

__device__ __forceinline__ void async_wait0() {
#if __has_builtin(__builtin_amdgcn_s_wait_asynccnt)
  __builtin_amdgcn_s_wait_asynccnt(0);
#else
  asm volatile("s_wait_asynccnt 0" ::: "memory");
#endif
}

// ---------- staging ----------

// contiguous global -> LDS A strip; async-to-LDS path on CDNA5 (16B per lane)
__device__ __forceinline__ void stage_a(f16* __restrict__ dst, const f16* __restrict__ src,
                                        int nelem, int tid, int nthr) {
  int n8 = nelem >> 3;
#if ASYNC_OK
  for (int i = tid; i < n8; i += nthr) {
    __builtin_amdgcn_global_load_async_to_lds_b128(
        (as1_v4i*)(void*)(src + i * 8),
        (as3_v4i*)(dst + i * 8), 0, 0);
  }
  async_wait0();
#else
  const uint4* s = (const uint4*)src;
  uint4* d = (uint4*)dst;
  for (int i = tid; i < n8; i += nthr) d[i] = s[i];
#endif
}

// Pack a 32 x ncols f16 slab (row-major, row stride ldg) into paired form:
// P[k2*stride + c] = src[2k2][c] | (src[2k2+1][c] << 16), k2 in [0,16)
__device__ __forceinline__ void stage_pairs(unsigned* __restrict__ P, int stride,
                                            const f16* __restrict__ src, int ldg, int ncols,
                                            int tid, int nthr) {
  int nc4 = ncols >> 2;
  int units = 16 * nc4;
  for (int u = tid; u < units; u += nthr) {
    int k2 = u / nc4, c0 = (u % nc4) << 2;
    uint2 a = *(const uint2*)(src + (2 * k2) * ldg + c0);
    uint2 b = *(const uint2*)(src + (2 * k2 + 1) * ldg + c0);
    uint4 o;
    o.x = (a.x & 0xffffu) | (b.x << 16);
    o.y = (a.x >> 16)     | (b.x & 0xffff0000u);
    o.z = (a.y & 0xffffu) | (b.y << 16);
    o.w = (a.y >> 16)     | (b.y & 0xffff0000u);
    *(uint4*)(P + k2 * stride + c0) = o;
  }
}

// Same, with row guard (rows >= rows_valid read as zero)
__device__ __forceinline__ void stage_pairs_guard(unsigned* __restrict__ P, int stride,
                                                  const f16* __restrict__ src, int ldg, int ncols,
                                                  int rows_valid, int tid, int nthr) {
  int nc4 = ncols >> 2;
  int units = 16 * nc4;
  for (int u = tid; u < units; u += nthr) {
    int k2 = u / nc4, c0 = (u % nc4) << 2;
    uint2 a = make_uint2(0u, 0u), b = make_uint2(0u, 0u);
    if (2 * k2 < rows_valid)     a = *(const uint2*)(src + (2 * k2) * ldg + c0);
    if (2 * k2 + 1 < rows_valid) b = *(const uint2*)(src + (2 * k2 + 1) * ldg + c0);
    uint4 o;
    o.x = (a.x & 0xffffu) | (b.x << 16);
    o.y = (a.x >> 16)     | (b.x & 0xffff0000u);
    o.z = (a.y & 0xffffu) | (b.y << 16);
    o.w = (a.y >> 16)     | (b.y & 0xffff0000u);
    *(uint4*)(P + k2 * stride + c0) = o;
  }
}

// ---------- WMMA fragment loaders (ISA 7.12.2 wave32 layouts) ----------

// A 16x32 f16 fragment from row-major LDS tile: pairs row-contiguous -> merges to 2x ds_load_b128
__device__ __forceinline__ v16h lds_a(const f16* __restrict__ A, int lda, int lane) {
  int m = lane & 15, h = lane >> 4;
  const f16* row = A + m * lda;
  v16h r;
#pragma unroll
  for (int j = 0; j < 8; ++j) {
    int kb = 2 * j + ((j >= 4) ? 8 : 0) + 8 * h;
    r[2 * j]     = row[kb];
    r[2 * j + 1] = row[kb + 1];
  }
  return r;
}

// B 32x16 fragment from paired slab: VGPR j <- P[(j + (j>=4?4:0) + 4h)*stride + n]  (8x ds_load_b32)
__device__ __forceinline__ v16h frag_b_p(const unsigned* __restrict__ P, int stride, int lane) {
  int n = lane & 15, h = lane >> 4;
  v8u u;
#pragma unroll
  for (int j = 0; j < 8; ++j) u[j] = P[(j + ((j >= 4) ? 4 : 0) + 4 * h) * stride + n];
  return __builtin_bit_cast(v16h, u);
}

// A^T 16x32 fragment (tile stored as k-major pairs): VGPR j <- P[(j + (j>=4?4:0) + 4h)*stride + m]
__device__ __forceinline__ v16h frag_at_p(const unsigned* __restrict__ P, int stride, int lane) {
  int m = lane & 15, h = lane >> 4;
  v8u u;
#pragma unroll
  for (int j = 0; j < 8; ++j) {
    int k2 = j + ((j >= 4) ? 4 : 0) + 4 * h;
    u[j] = P[k2 * stride + m];
  }
  return __builtin_bit_cast(v16h, u);
}

// ---------- small prep kernels ----------

__global__ void k_cvt(const float* __restrict__ s, f16* __restrict__ d, int n) {
  int i = blockIdx.x * blockDim.x + threadIdx.x;
  if (i < n) d[i] = (f16)s[i];
}

// x [B,T,N,D] -> Xg f16 [(b*D+d)*N+n, T]  and  xrh f16 [b*NP+n, T*D]
__global__ void k_pack(const float* __restrict__ x, f16* __restrict__ Xg, f16* __restrict__ xrh) {
  int i = blockIdx.x * blockDim.x + threadIdx.x;
  if (i >= CB * CT * CN * CD) return;
  int d = i & 1;
  int v = i >> 1;
  int n = v % CN;
  int bt = v / CN;
  int t = bt % CT;
  int b = bt / CT;
  float val = x[i];
  Xg[((b * CD + d) * CN + n) * CT + t] = (f16)val;
  xrh[(b * NP + n) * DM + t * CD + d]  = (f16)val;
}

// ---------- GCN GEMM 1: H1 = relu(diag * (Xg @ w1 + b1)),  [160000 x 256] ----------
__global__ void __launch_bounds__(256) k_gcn1(const f16* __restrict__ Xg, const f16* __restrict__ w1h,
                                              const float* __restrict__ b1, const float* __restrict__ graph,
                                              f16* __restrict__ H1) {
  __shared__ __align__(16) f16 As[16 * CT];
  __shared__ __align__(16) unsigned Bp[16 * (CH + 4)];
  constexpr int BS = CH + 4;
  int mt = blockIdx.x;
  int tid = threadIdx.x, wv = tid >> 5, lane = tid & 31;
  stage_a(As, Xg + mt * 16 * CT, 16 * CT, tid, 256);
  v8f acc[2] = { vzero(), vzero() };
  for (int ks = 0; ks < CT / 32; ++ks) {
    __syncthreads();
    stage_pairs(Bp, BS, w1h + ks * 32 * CH, CH, CH, tid, 256);
    __syncthreads();
    v16h a = lds_a(As + ks * 32, CT, lane);
#pragma unroll
    for (int ct = 0; ct < 2; ++ct) {
      v16h bf = frag_b_p(Bp + (wv * 2 + ct) * 16, BS, lane);
      acc[ct] = WMMA_F16(a, bf, acc[ct]);
    }
  }
  int nl = lane & 15, h = lane >> 4;
#pragma unroll
  for (int ct = 0; ct < 2; ++ct) {
    int col = (wv * 2 + ct) * 16 + nl;
    float bs = b1[col];
#pragma unroll
    for (int i = 0; i < 8; ++i) {
      int r = mt * 16 + i + 8 * h;
      int nn = r % CN;
      float dg = graph[nn * CN + nn];
      float v = (acc[ct][i] + bs) * dg;
      v = v > 0.f ? v : 0.f;
      H1[r * CH + col] = (f16)v;
    }
  }
}

// ---------- GCN GEMM 2: out = relu(diag * (H1 @ w2 + b2)) + x  -> d_out [B,T,N,D] ----------
__global__ void __launch_bounds__(192) k_gcn2(const f16* __restrict__ H1, const f16* __restrict__ w2h,
                                              const float* __restrict__ b2, const float* __restrict__ graph,
                                              const float* __restrict__ x, float* __restrict__ outp) {
  __shared__ __align__(16) f16 As[16 * CH];
  __shared__ __align__(16) unsigned Bp[16 * (CT + 4)];
  constexpr int BS = CT + 4;
  int mt = blockIdx.x;
  int tid = threadIdx.x, wv = tid >> 5, lane = tid & 31;
  stage_a(As, H1 + mt * 16 * CH, 16 * CH, tid, 192);
  v8f acc = vzero();
  for (int ks = 0; ks < CH / 32; ++ks) {
    __syncthreads();
    stage_pairs(Bp, BS, w2h + ks * 32 * CT, CT, CT, tid, 192);
    __syncthreads();
    v16h a  = lds_a(As + ks * 32, CH, lane);
    v16h bf = frag_b_p(Bp + wv * 16, BS, lane);
    acc = WMMA_F16(a, bf, acc);
  }
  int nl = lane & 15, h = lane >> 4;
  int col = wv * 16 + nl;                       // t
  float bs = b2[col];
#pragma unroll
  for (int i = 0; i < 8; ++i) {
    int r = mt * 16 + i + 8 * h;                // (b*D+d)*N+n
    int nn = r % CN;
    int bd = r / CN;
    int dd = bd % CD, bb = bd / CD;
    float dg = graph[nn * CN + nn];
    float v = (acc[i] + bs) * dg;
    v = v > 0.f ? v : 0.f;
    int xi = ((bb * CT + col) * CN + nn) * CD + dd;
    outp[xi] = v + x[xi];                       // gcn + residual x
  }
}

// ---------- QKV projections: mode 0/1 -> squared-relu + sin/cos split; mode 2 -> plain v ----------
__global__ void __launch_bounds__(128) k_qkv(const f16* __restrict__ xrh, const f16* __restrict__ Wh,
                                             const float* __restrict__ bias, f16* __restrict__ outp, int mode) {
  __shared__ __align__(16) f16 As[16 * DM];
  __shared__ __align__(16) unsigned Bp[16 * (DM + 4)];
  constexpr int BS = DM + 4;
  int mt = blockIdx.x, tid = threadIdx.x, wv = tid >> 5, lane = tid & 31;
  int n0 = (mt % MT_B) * 16, r0 = mt * 16;
  stage_a(As, xrh + r0 * DM, 16 * DM, tid, 128);
  v8f acc[3] = { vzero(), vzero(), vzero() };
  for (int ks = 0; ks < DM / 32; ++ks) {
    __syncthreads();
    stage_pairs(Bp, BS, Wh + ks * 32 * DM, DM, DM, tid, 128);
    __syncthreads();
    v16h a = lds_a(As + ks * 32, DM, lane);
#pragma unroll
    for (int ct = 0; ct < 3; ++ct) {
      v16h bf = frag_b_p(Bp + (wv * 3 + ct) * 16, BS, lane);
      acc[ct] = WMMA_F16(a, bf, acc[ct]);
    }
  }
  int nl = lane & 15, h = lane >> 4;
#pragma unroll
  for (int ct = 0; ct < 3; ++ct) {
    int col = (wv * 3 + ct) * 16 + nl;
    float bs = bias[col];
#pragma unroll
    for (int i = 0; i < 8; ++i) {
      int n = n0 + i + 8 * h;
      int r = r0 + i + 8 * h;
      float v = acc[ct][i] + bs;
      if (mode == 2) {
        outp[r * DM + col] = (n < CN) ? (f16)v : (f16)0.f;
      } else {
        float s = (v > 0.f && n < CN) ? v * v : 0.f;   // relu(q)*q
        float idx = PIH * (float)(n + 1) * (1.0f / CN);
        outp[r * DM2 + col]      = (f16)(s * __sinf(idx));
        outp[r * DM2 + DM + col] = (f16)(s * __cosf(idx));
      }
    }
  }
}

// ---------- ksum[b,c] = sum_n k_[b,n,c] ----------
__global__ void k_ksum(const f16* __restrict__ kh, float* __restrict__ ksum) {
  int b = blockIdx.x / 3;
  int c = (blockIdx.x % 3) * 128 + threadIdx.x;
  float acc = 0.f;
  for (int n = 0; n < CN; ++n) acc += (float)kh[(b * NP + n) * DM2 + c];
  ksum[b * DM2 + c] = acc;
}

// ---------- kv[b] = k_[b]^T @ v[b]  (384 x 192, K=5000), deterministic full-K per tile ----------
__global__ void __launch_bounds__(128) k_kv(const f16* __restrict__ kh, const f16* __restrict__ vh,
                                            f16* __restrict__ kvh) {
  __shared__ __align__(16) unsigned Kp[16 * 20];   // A^T pairs: 32 nodes x 16 feats
  __shared__ __align__(16) unsigned Vp[16 * 68];   // B pairs:   32 nodes x 64 v-cols
  int bid = blockIdx.x;
  int b = bid / 72, rem = bid % 72, mtile = rem / 3, ng = rem % 3;
  int tid = threadIdx.x, wv = tid >> 5, lane = tid & 31;
  v8f acc = vzero();
  for (int k0 = 0; k0 < CN; k0 += 32) {
    __syncthreads();
    stage_pairs_guard(Kp, 20, kh + (b * NP + k0) * DM2 + mtile * 16, DM2, 16, CN - k0, tid, 128);
    stage_pairs_guard(Vp, 68, vh + (b * NP + k0) * DM + ng * 64,     DM,  64, CN - k0, tid, 128);
    __syncthreads();
    v16h a  = frag_at_p(Kp, 20, lane);
    v16h bf = frag_b_p(Vp + wv * 16, 68, lane);
    acc = WMMA_F16(a, bf, acc);
  }
  int nl = lane & 15, h = lane >> 4;
  int col = ng * 64 + wv * 16 + nl;
#pragma unroll
  for (int i = 0; i < 8; ++i) {
    int row = mtile * 16 + i + 8 * h;
    kvh[(b * DM2 + row) * DM + col] = (f16)acc[i];
  }
}

// ---------- z[r] = 1/max(q_[r] . ksum[b], eps) ----------
__global__ void k_z(const f16* __restrict__ qh, const float* __restrict__ ksum, float* __restrict__ zbuf) {
  int r = blockIdx.x * blockDim.x + threadIdx.x;
  if (r >= RP) return;
  int b = r / NP, n = r % NP;
  if (n >= CN) { zbuf[r] = 0.f; return; }
  float acc = 0.f;
  for (int c = 0; c < DM2; ++c) acc += (float)qh[r * DM2 + c] * ksum[b * DM2 + c];
  zbuf[r] = 1.f / fmaxf(acc, FEPS);
}

// ---------- attn = (q_ @ kv[b]) * z + residual(xr), stored f16 for the wo GEMM ----------
__global__ void __launch_bounds__(128) k_attn(const f16* __restrict__ qh, const f16* __restrict__ kvh,
                                              const float* __restrict__ zbuf, const float* __restrict__ x,
                                              f16* __restrict__ attnh) {
  __shared__ __align__(16) f16 As[16 * DM2];
  __shared__ __align__(16) unsigned Bp[16 * (DM + 4)];
  constexpr int BS = DM + 4;
  int mt = blockIdx.x, tid = threadIdx.x, wv = tid >> 5, lane = tid & 31;
  int b = mt / MT_B, n0 = (mt % MT_B) * 16, r0 = mt * 16;
  stage_a(As, qh + r0 * DM2, 16 * DM2, tid, 128);
  v8f acc[3] = { vzero(), vzero(), vzero() };
  for (int ks = 0; ks < DM2 / 32; ++ks) {
    __syncthreads();
    stage_pairs(Bp, BS, kvh + (b * DM2 + ks * 32) * DM, DM, DM, tid, 128);
    __syncthreads();
    v16h a = lds_a(As + ks * 32, DM2, lane);
#pragma unroll
    for (int ct = 0; ct < 3; ++ct) {
      v16h bf = frag_b_p(Bp + (wv * 3 + ct) * 16, BS, lane);
      acc[ct] = WMMA_F16(a, bf, acc[ct]);
    }
  }
  int nl = lane & 15, h = lane >> 4;
#pragma unroll
  for (int ct = 0; ct < 3; ++ct) {
    int col = (wv * 3 + ct) * 16 + nl;
    int t = col >> 1, dd = col & 1;
#pragma unroll
    for (int i = 0; i < 8; ++i) {
      int n = n0 + i + 8 * h, r = r0 + i + 8 * h;
      f16 o = (f16)0.f;
      if (n < CN) {
        float res = x[((b * CT + t) * CN + n) * CD + dd];
        o = (f16)(acc[ct][i] * zbuf[r] + res);
      }
      attnh[r * DM + col] = o;
    }
  }
}

// ---------- out += (attn+res) @ wo + bo, scattered back to [B,T,N,D] ----------
__global__ void __launch_bounds__(128) k_out(const f16* __restrict__ attnh, const f16* __restrict__ woh,
                                             const float* __restrict__ bo, float* __restrict__ outp) {
  __shared__ __align__(16) f16 As[16 * DM];
  __shared__ __align__(16) unsigned Bp[16 * (DM + 4)];
  constexpr int BS = DM + 4;
  int mt = blockIdx.x, tid = threadIdx.x, wv = tid >> 5, lane = tid & 31;
  int b = mt / MT_B, n0 = (mt % MT_B) * 16, r0 = mt * 16;
  stage_a(As, attnh + r0 * DM, 16 * DM, tid, 128);
  v8f acc[3] = { vzero(), vzero(), vzero() };
  for (int ks = 0; ks < DM / 32; ++ks) {
    __syncthreads();
    stage_pairs(Bp, BS, woh + ks * 32 * DM, DM, DM, tid, 128);
    __syncthreads();
    v16h a = lds_a(As + ks * 32, DM, lane);
#pragma unroll
    for (int ct = 0; ct < 3; ++ct) {
      v16h bf = frag_b_p(Bp + (wv * 3 + ct) * 16, BS, lane);
      acc[ct] = WMMA_F16(a, bf, acc[ct]);
    }
  }
  int nl = lane & 15, h = lane >> 4;
#pragma unroll
  for (int ct = 0; ct < 3; ++ct) {
    int col = (wv * 3 + ct) * 16 + nl;
    int t = col >> 1, dd = col & 1;
    float bs = bo[col];
#pragma unroll
    for (int i = 0; i < 8; ++i) {
      int n = n0 + i + 8 * h;
      if (n < CN) {
        int oi = ((b * CT + t) * CN + n) * CD + dd;
        outp[oi] += acc[ct][i] + bs;           // d_out already holds gcn+x
      }
    }
  }
}

extern "C" void kernel_launch(void* const* d_in, const int* in_sizes, int n_in,
                              void* d_out, int out_size, void* d_ws, size_t ws_size,
                              hipStream_t stream) {
  (void)in_sizes; (void)n_in; (void)out_size; (void)ws_size;
  const float* x     = (const float*)d_in[0];
  const float* graph = (const float*)d_in[1];
  const float* w1 = (const float*)d_in[2];  const float* b1 = (const float*)d_in[3];
  const float* w2 = (const float*)d_in[4];  const float* b2 = (const float*)d_in[5];
  const float* wq = (const float*)d_in[6];  const float* bq = (const float*)d_in[7];
  const float* wk = (const float*)d_in[8];  const float* bk = (const float*)d_in[9];
  const float* wv = (const float*)d_in[10]; const float* bv = (const float*)d_in[11];
  const float* wo = (const float*)d_in[12]; const float* bo = (const float*)d_in[13];
  float* outp = (float*)d_out;

  char* base = (char*)d_ws;
  size_t cur = 0;
  auto take = [&](size_t elems, size_t esz) -> void* {
    void* p = base + cur;
    cur += (elems * esz + 255) & ~(size_t)255;
    return p;
  };
  f16* w1h = (f16*)take(CT * CH, 2);
  f16* w2h = (f16*)take(CH * CT, 2);
  f16* wqh = (f16*)take(DM * DM, 2);
  f16* wkh = (f16*)take(DM * DM, 2);
  f16* wvh = (f16*)take(DM * DM, 2);
  f16* woh = (f16*)take(DM * DM, 2);
  f16* Xg  = (f16*)take((size_t)R1 * CT, 2);
  f16* H1  = (f16*)take((size_t)R1 * CH, 2);
  f16* xrh = (f16*)take((size_t)RP * DM, 2);
  f16* qh  = (f16*)take((size_t)RP * DM2, 2);
  f16* kh  = (f16*)take((size_t)RP * DM2, 2);
  f16* vh  = (f16*)take((size_t)RP * DM, 2);
  f16* ath = (f16*)take((size_t)RP * DM, 2);
  f16* kvh = (f16*)take((size_t)CB * DM2 * DM, 2);
  float* ksum = (float*)take(CB * DM2, 4);
  float* zbuf = (float*)take(RP, 4);

  // weight f32 -> f16
  k_cvt<<<(CT * CH + 255) / 256, 256, 0, stream>>>(w1, w1h, CT * CH);
  k_cvt<<<(CH * CT + 255) / 256, 256, 0, stream>>>(w2, w2h, CH * CT);
  k_cvt<<<(DM * DM + 255) / 256, 256, 0, stream>>>(wq, wqh, DM * DM);
  k_cvt<<<(DM * DM + 255) / 256, 256, 0, stream>>>(wk, wkh, DM * DM);
  k_cvt<<<(DM * DM + 255) / 256, 256, 0, stream>>>(wv, wvh, DM * DM);
  k_cvt<<<(DM * DM + 255) / 256, 256, 0, stream>>>(wo, woh, DM * DM);

  // pack activations (both GCN-row and attention-row layouts)
  k_pack<<<(CB * CT * CN * CD + 255) / 256, 256, 0, stream>>>(x, Xg, xrh);

  // GCN path -> d_out = gcn(x) + x
  k_gcn1<<<R1 / 16, 256, 0, stream>>>(Xg, w1h, b1, graph, H1);
  k_gcn2<<<R1 / 16, 192, 0, stream>>>(H1, w2h, b2, graph, x, outp);

  // attention path
  k_qkv<<<CB * MT_B, 128, 0, stream>>>(xrh, wqh, bq, qh, 0);
  k_qkv<<<CB * MT_B, 128, 0, stream>>>(xrh, wkh, bk, kh, 1);
  k_qkv<<<CB * MT_B, 128, 0, stream>>>(xrh, wvh, bv, vh, 2);
  k_ksum<<<CB * 3, 128, 0, stream>>>(kh, ksum);
  k_kv<<<CB * 24 * 3, 128, 0, stream>>>(kh, vh, kvh);
  k_z<<<RP / 256, 256, 0, stream>>>(qh, ksum, zbuf);
  k_attn<<<CB * MT_B, 128, 0, stream>>>(qh, kvh, zbuf, x, ath);
  k_out<<<CB * MT_B, 128, 0, stream>>>(ath, woh, bo, outp);
}
